// GAE_IR_79499844649397
// MI455X (gfx1250) — compile-verified, matrix operand order checked by reading, not run on hardware
//
#include <hip/hip_runtime.h>
#include <hip/hip_bf16.h>

typedef __bf16 bf16_t;
typedef __attribute__((ext_vector_type(16))) __bf16 v16bf;
typedef __attribute__((ext_vector_type(8)))  float  v8f;
typedef __attribute__((ext_vector_type(4)))  unsigned int vuint4;

#define BM 64
#define BN 128
#define BK 32
#define APAD 8   // +16B per row -> 80B row stride for conflict-spread b128 reads

union V16 { vuint4 q[2]; v16bf v; };

#define WMMA_BF16(a, b, c) \
  __builtin_amdgcn_wmma_f32_16x16x32_bf16(false, (a), false, (b), (short)0, (c), false, false)

// ---------------------------------------------------------------------------
// GEMM (TN):  C[M,N] (+)= act( A[M,K] @ Bt[N,K]^T + bias )
// A bf16 row-major (lda=K), Bt bf16 row-major [N][K] (pre-transposed weight).
// Staging: global_load_async_to_lds_b128 (ASYNCcnt) + s_wait_asynccnt.
// Compute: per-wave 32x32 tile = 2x2 v_wmma_f32_16x16x32_bf16, fed by
// ds_load_b128 fragment loads. Optional f32 C (beta accumulate + row scatter)
// and/or bf16 copy Cb (feeds next GEMM's A operand).
// ---------------------------------------------------------------------------
__global__ __launch_bounds__(256) void gemm_bf16_tn(
    const bf16_t* __restrict__ A,
    const bf16_t* __restrict__ Bt,
    const float* __restrict__ bias,
    float* __restrict__ C, int ldc,
    bf16_t* __restrict__ Cb, int ldcb,
    const int* __restrict__ row_map,
    int M, int N, int K, int beta, int relu)
{
  __shared__ __align__(16) bf16_t As[BM][BK + APAD];
  __shared__ __align__(16) bf16_t Bs[BN][BK + APAD];

  const int tid  = threadIdx.x;
  const int bm   = blockIdx.x * BM;
  const int bn   = blockIdx.y * BN;
  const int wave = tid >> 5;
  const int lane = tid & 31;
  const int wrow = (wave >> 2) * 32;   // 0 / 32
  const int wcol = (wave & 3) * 32;    // 0 / 32 / 64 / 96
  const int hf   = lane >> 4;
  const int r    = lane & 15;

  // staging: 16B (8 bf16) chunks; every K here is a multiple of 8
  const int s_row = tid >> 2;          // 0..63
  const int s_c8  = (tid & 3) * 8;     // 0/8/16/24

  const vuint4 zero4 = {0u, 0u, 0u, 0u};
  v8f acc[2][2] = {};

  for (int k0 = 0; k0 < K; k0 += BK) {
    // ---- A tile 64x32: exactly one chunk per thread, async global->LDS
    {
      int gm = bm + s_row, gk = k0 + s_c8;
      void* ldsp = (void*)&As[s_row][s_c8];
      unsigned lds32 = (unsigned)(size_t)ldsp;
      if (gm < M && gk < K) {
        const bf16_t* gp = A + (size_t)gm * K + gk;
        asm volatile("global_load_async_to_lds_b128 %0, %1, off"
                     :: "v"(lds32), "v"(gp) : "memory");
      } else {
        *(vuint4*)ldsp = zero4;   // zero-fill OOB chunk (no addr overlap w/ async)
      }
    }
    // ---- B tile 128x32: two chunks per thread
#pragma unroll
    for (int t = 0; t < 2; ++t) {
      int i = tid + t * 256;
      int row = i >> 2, c8 = (i & 3) * 8;
      int gn = bn + row, gk = k0 + c8;
      void* ldsp = (void*)&Bs[row][c8];
      unsigned lds32 = (unsigned)(size_t)ldsp;
      if (gn < N && gk < K) {
        const bf16_t* gp = Bt + (size_t)gn * K + gk;
        asm volatile("global_load_async_to_lds_b128 %0, %1, off"
                     :: "v"(lds32), "v"(gp) : "memory");
      } else {
        *(vuint4*)ldsp = zero4;
      }
    }
    // speculative prefetch of next K tile (-> global_prefetch_b8)
    if (k0 + BK < K) {
      int gm = bm + s_row;
      if (gm < M) __builtin_prefetch(A + (size_t)gm * K + (k0 + BK) + s_c8, 0, 1);
    }
    asm volatile("s_wait_asynccnt 0x0" ::: "memory");
    __syncthreads();

    // ---- fragments: lane(hf,r) needs K runs [8*hf,+8) and [16+8*hf,+8)
    V16 a0, a1, b0, b1;
    const bf16_t* ar0 = &As[wrow + r][0];
    const bf16_t* ar1 = &As[wrow + 16 + r][0];
    const bf16_t* br0 = &Bs[wcol + r][0];
    const bf16_t* br1 = &Bs[wcol + 16 + r][0];
    a0.q[0] = *(const vuint4*)(ar0 + 8 * hf);
    a0.q[1] = *(const vuint4*)(ar0 + 16 + 8 * hf);
    a1.q[0] = *(const vuint4*)(ar1 + 8 * hf);
    a1.q[1] = *(const vuint4*)(ar1 + 16 + 8 * hf);
    b0.q[0] = *(const vuint4*)(br0 + 8 * hf);
    b0.q[1] = *(const vuint4*)(br0 + 16 + 8 * hf);
    b1.q[0] = *(const vuint4*)(br1 + 8 * hf);
    b1.q[1] = *(const vuint4*)(br1 + 16 + 8 * hf);

    acc[0][0] = WMMA_BF16(a0.v, b0.v, acc[0][0]);
    acc[0][1] = WMMA_BF16(a0.v, b1.v, acc[0][1]);
    acc[1][0] = WMMA_BF16(a1.v, b0.v, acc[1][0]);
    acc[1][1] = WMMA_BF16(a1.v, b1.v, acc[1][1]);
    __syncthreads();
  }

  // ---- epilogue: C/D layout M = v + 8*hf, N = r
#pragma unroll
  for (int i = 0; i < 2; ++i) {
#pragma unroll
    for (int j = 0; j < 2; ++j) {
#pragma unroll
      for (int v = 0; v < 8; ++v) {
        int gm = bm + wrow + i * 16 + v + 8 * hf;
        int gn = bn + wcol + j * 16 + r;
        if (gm < M && gn < N) {
          float val = acc[i][j][v];
          if (bias) val += bias[gn];
          int orow = row_map ? row_map[gm] : gm;
          if (beta && C) val += C[(size_t)orow * ldc + gn];
          if (relu) val = fmaxf(val, 0.f);
          if (C)  C[(size_t)orow * ldc + gn] = val;
          if (Cb) Cb[(size_t)orow * ldcb + gn] = (bf16_t)val;
        }
      }
    }
  }
}

// ---------------------------------------------------------------------------
// helpers
// ---------------------------------------------------------------------------
__global__ void k_zero_f32(float* __restrict__ p, size_t n) {
  size_t i = (size_t)blockIdx.x * blockDim.x + threadIdx.x;
  if (i < n) p[i] = 0.f;
}

__global__ void k_f32_to_bf16(const float* __restrict__ x, bf16_t* __restrict__ y, size_t n) {
  size_t i = (size_t)blockIdx.x * blockDim.x + threadIdx.x;
  if (i < n) y[i] = (bf16_t)x[i];
}

// Wt[n][k] = W[k][n]  (f32 -> bf16, writes coalesced along k)
__global__ void k_w_transpose_bf16(const float* __restrict__ W, int K, int N,
                                   bf16_t* __restrict__ Wt) {
  size_t i = (size_t)blockIdx.x * blockDim.x + threadIdx.x;
  if (i >= (size_t)K * N) return;
  int n = (int)(i / K), k = (int)(i % K);
  Wt[i] = (bf16_t)W[(size_t)k * N + n];
}

__global__ void k_edge_count(const int* __restrict__ dst, int E, float* __restrict__ cnt) {
  int i = blockIdx.x * blockDim.x + threadIdx.x;
  if (i < E) atomicAdd(&cnt[dst[i]], 1.f);
}

__global__ void k_edge_agg(const int* __restrict__ src, const int* __restrict__ dst, int E,
                           const float* __restrict__ X, int D, float* __restrict__ agg) {
  size_t i = (size_t)blockIdx.x * blockDim.x + threadIdx.x;
  if (i < (size_t)E * D) {
    int e = (int)(i / D), f = (int)(i % D);
    atomicAdd(&agg[(size_t)dst[e] * D + f], X[(size_t)src[e] * D + f]);
  }
}

// agg /= max(cnt,1), emit both f32 and bf16
__global__ void k_div_cnt_bf16(float* __restrict__ agg, const float* __restrict__ cnt,
                               int Nn, int D, bf16_t* __restrict__ agg_b) {
  size_t i = (size_t)blockIdx.x * blockDim.x + threadIdx.x;
  if (i < (size_t)Nn * D) {
    float v = agg[i] / fmaxf(cnt[i / D], 1.f);
    agg[i] = v;
    agg_b[i] = (bf16_t)v;
  }
}

// pairs_in[p, 0:nd]=z[attr], pairs_in[p, nd:2nd]=z[nattrs+obj]  (bf16)
__global__ void k_build_pairs(const float* __restrict__ z,
                              const int* __restrict__ t_attr,
                              const int* __restrict__ neg_attr,
                              const int* __restrict__ obj_ids,
                              const int* __restrict__ nattrs_p,
                              int B, int Kneg, int nd,
                              bf16_t* __restrict__ out) {
  size_t idx = (size_t)blockIdx.x * blockDim.x + threadIdx.x;
  int W2 = 2 * nd;
  if (idx >= (size_t)B * (Kneg + 1) * W2) return;
  size_t p = idx / W2;
  int c = (int)(idx % W2);
  int b = (int)(p / (Kneg + 1));
  int j = (int)(p % (Kneg + 1));
  int nid, cc;
  if (c < nd) {
    nid = (j == 0) ? t_attr[b] : neg_attr[(size_t)b * Kneg + (j - 1)];
    cc = c;
  } else {
    nid = nattrs_p[0] + obj_ids[b];
    cc = c - nd;
  }
  out[idx] = (bf16_t)z[(size_t)nid * nd + cc];
}

// compo_in[p] = [ s_img[b] | pair_e[p] ]  (bf16)
__global__ void k_build_compo(const float* __restrict__ s_img,
                              const float* __restrict__ pair_e,
                              int B, int Kneg, int IMG, int EMB,
                              bf16_t* __restrict__ out) {
  size_t idx = (size_t)blockIdx.x * blockDim.x + threadIdx.x;
  int W2 = IMG + EMB;
  if (idx >= (size_t)B * (Kneg + 1) * W2) return;
  size_t p = idx / W2;
  int c = (int)(idx % W2);
  int b = (int)(p / (Kneg + 1));
  float v = (c < IMG) ? s_img[(size_t)b * IMG + c] : pair_e[p * EMB + (c - IMG)];
  out[idx] = (bf16_t)v;
}

__global__ void k_rowmap(int* __restrict__ map, int B, int Kneg) {
  int p = blockIdx.x * blockDim.x + threadIdx.x;
  if (p >= B * (Kneg + 1)) return;
  int b = p / (Kneg + 1), j = p % (Kneg + 1);
  map[p] = (j == 0) ? b : (2 * B + b * Kneg + (j - 1));
}

// in-place row-wise L2 normalize (eps 1e-12)
__global__ __launch_bounds__(256) void k_l2norm_rows(float* __restrict__ X, int D) {
  int row = blockIdx.x;
  float s = 0.f;
  for (int c = threadIdx.x; c < D; c += blockDim.x) {
    float v = X[(size_t)row * D + c];
    s += v * v;
  }
  __shared__ float red[256];
  red[threadIdx.x] = s;
  __syncthreads();
  for (int off = 128; off > 0; off >>= 1) {
    if ((int)threadIdx.x < off) red[threadIdx.x] += red[threadIdx.x + off];
    __syncthreads();
  }
  float inv = 1.f / fmaxf(sqrtf(red[0]), 1e-12f);
  for (int c = threadIdx.x; c < D; c += blockDim.x)
    X[(size_t)row * D + c] *= inv;
}

// ---------------------------------------------------------------------------
static inline dim3 gemm_grid(int M, int N) {
  return dim3((M + BM - 1) / BM, (N + BN - 1) / BN);
}
static inline int blks(size_t n, int t = 256) { return (int)((n + t - 1) / t); }

extern "C" void kernel_launch(void* const* d_in, const int* in_sizes, int n_in,
                              void* d_out, int out_size, void* d_ws, size_t ws_size,
                              hipStream_t stream) {
  // ---- inputs (setup_inputs order) ----
  const float* nodes = (const float*)d_in[0];
  const float* s_img = (const float*)d_in[1];
  const float* t_img = (const float*)d_in[2];
  const float *Wl1 = (const float*)d_in[3], *Wr1 = (const float*)d_in[4], *b1 = (const float*)d_in[5];
  const float *Wl2 = (const float*)d_in[6], *Wr2 = (const float*)d_in[7], *b2 = (const float*)d_in[8];
  const float *Wp1 = (const float*)d_in[9],  *bp1 = (const float*)d_in[10];
  const float *Wp2 = (const float*)d_in[11], *bp2 = (const float*)d_in[12];
  const float *Wi1 = (const float*)d_in[13], *bi1 = (const float*)d_in[14];
  const float *Wi2 = (const float*)d_in[15], *bi2 = (const float*)d_in[16];
  const float *Wi3 = (const float*)d_in[17], *bi3 = (const float*)d_in[18];
  const float *Wc1 = (const float*)d_in[19], *bc1 = (const float*)d_in[20];
  const float *Wc2 = (const float*)d_in[21], *bc2 = (const float*)d_in[22];
  const float *Wc3 = (const float*)d_in[23], *bc3 = (const float*)d_in[24];
  const int* edge_src = (const int*)d_in[25];
  const int* edge_dst = (const int*)d_in[26];
  const int* t_attr_ids = (const int*)d_in[27];
  const int* obj_ids = (const int*)d_in[28];
  const int* neg_attr_ids = (const int*)d_in[29];
  const int* nattrs_p = (const int*)d_in[30];

  // ---- dims from in_sizes ----
  const int B    = in_sizes[27];
  const int E    = in_sizes[25];
  const int HID  = in_sizes[5];
  const int ND   = in_sizes[8];
  const int P1   = in_sizes[10];
  const int EMB  = in_sizes[12];
  const int IMG  = in_sizes[1] / B;
  const int FEAT = in_sizes[3] / HID;
  const int Nn   = in_sizes[0] / FEAT;
  const int Kneg = in_sizes[29] / B;
  const int IH1  = in_sizes[14];
  const int IH2  = in_sizes[16];
  const int CH1  = in_sizes[20];
  const int CH2  = in_sizes[22];
  const int P    = B * (Kneg + 1);

  // ---- carve workspace ----
  uint8_t* w = (uint8_t*)d_ws;
  auto carve = [&](size_t bytes) -> void* {
    void* p = (void*)w;
    w += (bytes + 255) & ~(size_t)255;
    return p;
  };
  bf16_t* Wl1t = (bf16_t*)carve((size_t)FEAT * HID * 2);
  bf16_t* Wr1t = (bf16_t*)carve((size_t)FEAT * HID * 2);
  bf16_t* Wl2t = (bf16_t*)carve((size_t)HID * ND * 2);
  bf16_t* Wr2t = (bf16_t*)carve((size_t)HID * ND * 2);
  bf16_t* Wp1t = (bf16_t*)carve((size_t)2 * ND * P1 * 2);
  bf16_t* Wp2t = (bf16_t*)carve((size_t)P1 * EMB * 2);
  bf16_t* Wi1t = (bf16_t*)carve((size_t)IMG * IH1 * 2);
  bf16_t* Wi2t = (bf16_t*)carve((size_t)IH1 * IH2 * 2);
  bf16_t* Wi3t = (bf16_t*)carve((size_t)IH2 * EMB * 2);
  bf16_t* Wc1t = (bf16_t*)carve((size_t)(IMG + EMB) * CH1 * 2);
  bf16_t* Wc2t = (bf16_t*)carve((size_t)CH1 * CH2 * 2);
  bf16_t* Wc3t = (bf16_t*)carve((size_t)CH2 * EMB * 2);
  float*  cnt     = (float*)carve((size_t)Nn * 4);
  float*  agg     = (float*)carve((size_t)Nn * HID * 4);
  bf16_t* agg_b   = (bf16_t*)carve((size_t)Nn * HID * 2);
  bf16_t* nodes_b = (bf16_t*)carve((size_t)Nn * FEAT * 2);
  float*  h       = (float*)carve((size_t)Nn * HID * 4);
  bf16_t* h_b     = (bf16_t*)carve((size_t)Nn * HID * 2);
  float*  z       = (float*)carve((size_t)Nn * ND * 4);
  bf16_t* pairs_in = (bf16_t*)carve((size_t)P * 2 * ND * 2);
  bf16_t* pair_h_b = (bf16_t*)carve((size_t)P * P1 * 2);
  float*  pair_e   = (float*)carve((size_t)P * EMB * 4);
  bf16_t* compo_in = (bf16_t*)carve((size_t)P * (IMG + EMB) * 2);
  bf16_t* c1_b     = pairs_in;   // reuse (pairs_in consumed by pair_h GEMM)
  bf16_t* c2_b     = pair_h_b;   // reuse (pair_h consumed by pair_e GEMM)
  bf16_t* timg_b = (bf16_t*)carve((size_t)B * IMG * 2);
  bf16_t* t1_b   = (bf16_t*)carve((size_t)B * IH1 * 2);
  bf16_t* t2_b   = (bf16_t*)carve((size_t)B * IH2 * 2);
  int*    rowmap = (int*)carve((size_t)P * 4);
  (void)ws_size; (void)n_in; (void)out_size;

  float* out = (float*)d_out;

  // ---- pre-transpose weights to bf16 [N][K] ----
  k_w_transpose_bf16<<<blks((size_t)FEAT * HID), 256, 0, stream>>>(Wl1, FEAT, HID, Wl1t);
  k_w_transpose_bf16<<<blks((size_t)FEAT * HID), 256, 0, stream>>>(Wr1, FEAT, HID, Wr1t);
  k_w_transpose_bf16<<<blks((size_t)HID * ND), 256, 0, stream>>>(Wl2, HID, ND, Wl2t);
  k_w_transpose_bf16<<<blks((size_t)HID * ND), 256, 0, stream>>>(Wr2, HID, ND, Wr2t);
  k_w_transpose_bf16<<<blks((size_t)2 * ND * P1), 256, 0, stream>>>(Wp1, 2 * ND, P1, Wp1t);
  k_w_transpose_bf16<<<blks((size_t)P1 * EMB), 256, 0, stream>>>(Wp2, P1, EMB, Wp2t);
  k_w_transpose_bf16<<<blks((size_t)IMG * IH1), 256, 0, stream>>>(Wi1, IMG, IH1, Wi1t);
  k_w_transpose_bf16<<<blks((size_t)IH1 * IH2), 256, 0, stream>>>(Wi2, IH1, IH2, Wi2t);
  k_w_transpose_bf16<<<blks((size_t)IH2 * EMB), 256, 0, stream>>>(Wi3, IH2, EMB, Wi3t);
  k_w_transpose_bf16<<<blks((size_t)(IMG + EMB) * CH1), 256, 0, stream>>>(Wc1, IMG + EMB, CH1, Wc1t);
  k_w_transpose_bf16<<<blks((size_t)CH1 * CH2), 256, 0, stream>>>(Wc2, CH1, CH2, Wc2t);
  k_w_transpose_bf16<<<blks((size_t)CH2 * EMB), 256, 0, stream>>>(Wc3, CH2, EMB, Wc3t);
  k_f32_to_bf16<<<blks((size_t)Nn * FEAT), 256, 0, stream>>>(nodes, nodes_b, (size_t)Nn * FEAT);
  k_f32_to_bf16<<<blks((size_t)B * IMG), 256, 0, stream>>>(t_img, timg_b, (size_t)B * IMG);

  // ---- SAGE layer 1: h = relu(mean_agg(nodes)@Wl1 + nodes@Wr1 + b1) ----
  k_zero_f32<<<blks(Nn), 256, 0, stream>>>(cnt, (size_t)Nn);
  k_zero_f32<<<blks((size_t)Nn * FEAT), 256, 0, stream>>>(agg, (size_t)Nn * FEAT);
  k_edge_count<<<blks(E), 256, 0, stream>>>(edge_dst, E, cnt);
  k_edge_agg<<<blks((size_t)E * FEAT), 256, 0, stream>>>(edge_src, edge_dst, E, nodes, FEAT, agg);
  k_div_cnt_bf16<<<blks((size_t)Nn * FEAT), 256, 0, stream>>>(agg, cnt, Nn, FEAT, agg_b);

  gemm_bf16_tn<<<gemm_grid(Nn, HID), 256, 0, stream>>>(
      agg_b, Wl1t, nullptr, h, HID, nullptr, 0, nullptr, Nn, HID, FEAT, 0, 0);
  gemm_bf16_tn<<<gemm_grid(Nn, HID), 256, 0, stream>>>(
      nodes_b, Wr1t, b1, h, HID, h_b, HID, nullptr, Nn, HID, FEAT, 1, 1);

  // ---- SAGE layer 2: z = mean_agg(h)@Wl2 + h@Wr2 + b2 ----
  k_zero_f32<<<blks((size_t)Nn * HID), 256, 0, stream>>>(agg, (size_t)Nn * HID);
  k_edge_agg<<<blks((size_t)E * HID), 256, 0, stream>>>(edge_src, edge_dst, E, h, HID, agg);
  k_div_cnt_bf16<<<blks((size_t)Nn * HID), 256, 0, stream>>>(agg, cnt, Nn, HID, agg_b);

  gemm_bf16_tn<<<gemm_grid(Nn, ND), 256, 0, stream>>>(
      agg_b, Wl2t, nullptr, z, ND, nullptr, 0, nullptr, Nn, ND, HID, 0, 0);
  gemm_bf16_tn<<<gemm_grid(Nn, ND), 256, 0, stream>>>(
      h_b, Wr2t, b2, z, ND, nullptr, 0, nullptr, Nn, ND, HID, 1, 0);

  // ---- pair batch: row 0 = positive, rows 1..K = negatives ----
  k_build_pairs<<<blks((size_t)P * 2 * ND), 256, 0, stream>>>(
      z, t_attr_ids, neg_attr_ids, obj_ids, nattrs_p, B, Kneg, ND, pairs_in);

  // pair_fc
  gemm_bf16_tn<<<gemm_grid(P, P1), 256, 0, stream>>>(
      pairs_in, Wp1t, bp1, nullptr, 0, pair_h_b, P1, nullptr, P, P1, 2 * ND, 0, 1);
  gemm_bf16_tn<<<gemm_grid(P, EMB), 256, 0, stream>>>(
      pair_h_b, Wp2t, bp2, pair_e, EMB, nullptr, 0, nullptr, P, EMB, P1, 0, 0);
  k_l2norm_rows<<<P, 256, 0, stream>>>(pair_e, EMB);

  // compo_fc on [s_img | pair_emb]
  k_build_compo<<<blks((size_t)P * (IMG + EMB)), 256, 0, stream>>>(
      s_img, pair_e, B, Kneg, IMG, EMB, compo_in);

  gemm_bf16_tn<<<gemm_grid(P, CH1), 256, 0, stream>>>(
      compo_in, Wc1t, bc1, nullptr, 0, c1_b, CH1, nullptr, P, CH1, IMG + EMB, 0, 1);
  gemm_bf16_tn<<<gemm_grid(P, CH2), 256, 0, stream>>>(
      c1_b, Wc2t, bc2, nullptr, 0, c2_b, CH2, nullptr, P, CH2, CH1, 0, 1);

  // final compo layer scattered straight into d_out (theta + neg rows)
  k_rowmap<<<blks(P), 256, 0, stream>>>(rowmap, B, Kneg);
  gemm_bf16_tn<<<gemm_grid(P, EMB), 256, 0, stream>>>(
      c2_b, Wc3t, bc3, out, EMB, nullptr, 0, rowmap, P, EMB, CH2, 0, 0);

  // targets = img_fc(t_img) -> rows [B, 2B)
  gemm_bf16_tn<<<gemm_grid(B, IH1), 256, 0, stream>>>(
      timg_b, Wi1t, bi1, nullptr, 0, t1_b, IH1, nullptr, B, IH1, IMG, 0, 1);
  gemm_bf16_tn<<<gemm_grid(B, IH2), 256, 0, stream>>>(
      t1_b, Wi2t, bi2, nullptr, 0, t2_b, IH2, nullptr, B, IH2, IH1, 0, 1);
  gemm_bf16_tn<<<gemm_grid(B, EMB), 256, 0, stream>>>(
      t2_b, Wi3t, bi3, out + (size_t)B * EMB, EMB, nullptr, 0, nullptr, B, EMB, IH2, 0, 0);

  // one L2-normalize pass over every output row (theta, targets, neg)
  k_l2norm_rows<<<2 * B + B * Kneg, 256, 0, stream>>>(out, EMB);
}
